// IsotropicAttention_73083163509294
// MI455X (gfx1250) — matrix operand order repaired and checked by reading, not compiled, not run on hardware
//
#include <hip/hip_runtime.h>
#include <hip/hip_bf16.h>

// ---------------------------------------------------------------------------
// IsotropicAttention for MI455X (gfx1250, wave32, WMMA bf16 16x16x32)
// ---------------------------------------------------------------------------

typedef __attribute__((ext_vector_type(16))) __bf16        v16bf;
typedef __attribute__((ext_vector_type(8)))  float         v8f;
typedef __attribute__((ext_vector_type(8)))  unsigned int  v8u;

#define D_MODEL  1024
#define N_HEADS  16
#define HEAD_DIM 64
#define SEQ_T    2048

// round-to-nearest-even fp32 -> bf16 (upper 16 bits)
__device__ __forceinline__ unsigned f2bf1(float f) {
    unsigned u = __float_as_uint(f);
    return (u + 0x7FFFu + ((u >> 16) & 1u)) >> 16;
}
__device__ __forceinline__ unsigned packbf(float lo, float hi) {
    return f2bf1(lo) | (f2bf1(hi) << 16);
}

// A-fragment K index: lane half g, packed-pair register j (16-bit A 16x32 layout)
__device__ __forceinline__ int kA_of(int j, int g) {
    return 2 * (j & 3) + ((j & 4) ? 16 : 0) + 8 * g;
}
// B-fragment K index (16-bit B 32x16 layout, [N][K] staging)
__device__ __forceinline__ int kB_of(int j, int g) {
    return 2 * (j & 3) + ((j & 4) ? 16 : 0) + 8 * g;
}

union U8u { v8u v; unsigned u[8]; };
union U8f { v8f v; float f[8]; };

// ---------------------------------------------------------------------------
// Generic C = A(f32, MxK) @ W(f32, NxK)^T   with optional output forms:
//   Cf   : fp32 row-major MxN
//   Cbf  : bf16 row-major MxN
//   Cvt  : bf16 per-head transposed  Vt[b][h][d][t]   (for the PV GEMM's B)
// Block 256 threads = 8 waves, tile 64x64, K-step 32, bf16 WMMA w/ f32 acc.
// ---------------------------------------------------------------------------
__global__ __launch_bounds__(256)
void proj_gemm_kernel(const float* __restrict__ A, const float* __restrict__ W,
                      float* __restrict__ Cf, unsigned short* __restrict__ Cbf,
                      unsigned short* __restrict__ Cvt, int M, int N, int K)
{
    __shared__ unsigned ldsA[64 * 16]; // 64 rows x 32 bf16 (as 16 dwords/row)
    __shared__ unsigned ldsB[64 * 16];

    const int tid  = threadIdx.x;
    const int lane = tid & 31;
    const int wave = tid >> 5;
    const int m0   = blockIdx.y * 64;
    const int n0   = blockIdx.x * 64;
    const int wm   = (wave >> 1) * 16;   // wave sub-tile row base within block
    const int wn   = (wave & 1) * 32;    // wave sub-tile col base (two 16-wide tiles)
    const int ml   = lane & 15;
    const int g    = lane >> 4;

    const int srow = tid >> 2;           // staging: row 0..63
    const int scol = (tid & 3) * 8;      // staging: 8 consecutive floats

    v8f c0 = {}; v8f c1 = {};

    for (int k0 = 0; k0 < K; k0 += 32) {
        // stage X and W tiles, fp32 -> packed bf16 pairs
        {
            const float* pa = A + (size_t)(m0 + srow) * K + k0 + scol;
            const float* pw = W + (size_t)(n0 + srow) * K + k0 + scol;
            unsigned* da = &ldsA[srow * 16 + (scol >> 1)];
            unsigned* dw = &ldsB[srow * 16 + (scol >> 1)];
            if (k0 + 32 < K) { // prefetch next K-slab (global_prefetch_b8)
                __builtin_prefetch(pa + 32, 0, 3);
                __builtin_prefetch(pw + 32, 0, 3);
            }
#pragma unroll
            for (int e = 0; e < 4; ++e) {
                da[e] = packbf(pa[2 * e], pa[2 * e + 1]);
                dw[e] = packbf(pw[2 * e], pw[2 * e + 1]);
            }
        }
        __syncthreads();

        U8u au, b0u, b1u;
#pragma unroll
        for (int j = 0; j < 8; ++j) {
            au.u[j]  = ldsA[(wm + ml) * 16 + (kA_of(j, g) >> 1)];
            b0u.u[j] = ldsB[(wn + ml) * 16 + (kB_of(j, g) >> 1)];
            b1u.u[j] = ldsB[(wn + 16 + ml) * 16 + (kB_of(j, g) >> 1)];
        }
        v16bf a  = __builtin_bit_cast(v16bf, au.v);
        v16bf b0 = __builtin_bit_cast(v16bf, b0u.v);
        v16bf b1 = __builtin_bit_cast(v16bf, b1u.v);
        c0 = __builtin_amdgcn_wmma_f32_16x16x32_bf16(false, a, false, b0, (short)0, c0, false, false);
        c1 = __builtin_amdgcn_wmma_f32_16x16x32_bf16(false, a, false, b1, (short)0, c1, false, false);
        __syncthreads();
    }

    // C layout: lane l holds n = l&15, rows m = r + 8*(l>>4)
    U8f u0, u1; u0.v = c0; u1.v = c1;
#pragma unroll
    for (int r = 0; r < 8; ++r) {
        const int gm  = m0 + wm + r + 8 * g;
        const int gn0 = n0 + wn + ml;
        const int gn1 = gn0 + 16;
        const float v0 = u0.f[r], v1 = u1.f[r];
        if (Cf) {
            Cf[(size_t)gm * N + gn0] = v0;
            Cf[(size_t)gm * N + gn1] = v1;
        }
        if (Cbf) {
            Cbf[(size_t)gm * N + gn0] = (unsigned short)f2bf1(v0);
            Cbf[(size_t)gm * N + gn1] = (unsigned short)f2bf1(v1);
        }
        if (Cvt) { // Vt[b][h][d][t], t contiguous
            const int b = gm >> 11, t = gm & (SEQ_T - 1);
            const int h0 = gn0 >> 6, d0 = gn0 & 63;
            const int h1 = gn1 >> 6, d1 = gn1 & 63;
            Cvt[((size_t)(b * N_HEADS + h0) * HEAD_DIM + d0) * SEQ_T + t] = (unsigned short)f2bf1(v0);
            Cvt[((size_t)(b * N_HEADS + h1) * HEAD_DIM + d1) * SEQ_T + t] = (unsigned short)f2bf1(v1);
        }
    }
}

// ---------------------------------------------------------------------------
// axis[b,h,t] = sum_d nodes[b,t,h*64+d] * w0[h,d]
// ---------------------------------------------------------------------------
__global__ __launch_bounds__(256)
void axis_kernel(const float* __restrict__ nodes, const float* __restrict__ w0,
                 float* __restrict__ axisb)
{
    const int idx = blockIdx.x * blockDim.x + threadIdx.x;   // over B*H*T
    const int t  = idx & (SEQ_T - 1);
    const int bh = idx >> 11;
    const int h  = bh & (N_HEADS - 1);
    const int b  = bh >> 4;
    const float* nr = nodes + ((size_t)(b * SEQ_T + t)) * D_MODEL + h * HEAD_DIM;
    const float* wr = w0 + h * HEAD_DIM;
    float s = 0.f;
#pragma unroll 8
    for (int d = 0; d < HEAD_DIM; ++d) s += nr[d] * wr[d];
    axisb[(size_t)bh * SEQ_T + t] = s;
}

// ---------------------------------------------------------------------------
// Pass 1: one wave per 16-row tile per (b,h).
// scores = (n_i . n_j) * inv_sqrt * sigmoid((ax_i + ax_j) * inv_sqrt)
// Write modulated scores to the attn output region; keep per-lane online
// (max, sum-exp) stats per row, combined across the 16-lane half at the end.
// ---------------------------------------------------------------------------
__global__ __launch_bounds__(32)
void scores_pass1(const unsigned short* __restrict__ nbf, const float* __restrict__ axisb,
                  float* __restrict__ attn, float* __restrict__ rowmax,
                  float* __restrict__ rowsum)
{
    const int bh   = blockIdx.x >> 7;               // T/16 = 128 row tiles per head
    const int i0   = (blockIdx.x & 127) * 16;
    const int lane = threadIdx.x;
    const int ml   = lane & 15;
    const int g    = lane >> 4;
    const float inv_sqrt = 0.125f;                  // 1/sqrt(64)

    const int b = bh >> 4, h = bh & (N_HEADS - 1);
    const unsigned short* nb = nbf + (size_t)b * SEQ_T * D_MODEL + h * HEAD_DIM;
    const float* axf = axisb + (size_t)bh * SEQ_T;
    float* attnBase = attn + (size_t)bh * SEQ_T * SEQ_T;

    // Q fragments for this row tile (K = hd = 64 -> two 16x32 bf16 A-frags)
    U8u au0, au1;
    {
        const unsigned* arow = (const unsigned*)(nb + (size_t)(i0 + ml) * D_MODEL);
#pragma unroll
        for (int j = 0; j < 8; ++j) {
            au0.u[j] = arow[kA_of(j, g) >> 1];
            au1.u[j] = arow[(32 + kA_of(j, g)) >> 1];
        }
    }
    const v16bf aF0 = __builtin_bit_cast(v16bf, au0.v);
    const v16bf aF1 = __builtin_bit_cast(v16bf, au1.v);

    float ax_row[8];
#pragma unroll
    for (int r = 0; r < 8; ++r) ax_row[r] = axf[i0 + r + 8 * g];

    float rmax[8], rsum[8];
#pragma unroll
    for (int r = 0; r < 8; ++r) { rmax[r] = -3.0e38f; rsum[r] = 0.f; }

    for (int jt = 0; jt < SEQ_T / 16; ++jt) {
        const int j0 = jt * 16;
        const float ax_col = axf[j0 + ml];

        U8u bu0, bu1;
        const unsigned* brow = (const unsigned*)(nb + (size_t)(j0 + ml) * D_MODEL);
#pragma unroll
        for (int j = 0; j < 8; ++j) {
            bu0.u[j] = brow[kB_of(j, g) >> 1];
            bu1.u[j] = brow[(32 + kB_of(j, g)) >> 1];
        }
        v8f c = {};
        c = __builtin_amdgcn_wmma_f32_16x16x32_bf16(false, aF0, false,
                __builtin_bit_cast(v16bf, bu0.v), (short)0, c, false, false);
        c = __builtin_amdgcn_wmma_f32_16x16x32_bf16(false, aF1, false,
                __builtin_bit_cast(v16bf, bu1.v), (short)0, c, false, false);

        U8f cu; cu.v = c;
#pragma unroll
        for (int r = 0; r < 8; ++r) {
            float s = cu.f[r] * inv_sqrt;
            const float mz = (ax_row[r] + ax_col) * inv_sqrt;
            s *= 1.f / (1.f + __expf(-mz));                  // sigmoid modulation
            attnBase[(size_t)(i0 + r + 8 * g) * SEQ_T + j0 + ml] = s;
            const float nm = fmaxf(rmax[r], s);              // per-lane online stats
            rsum[r] = rsum[r] * __expf(rmax[r] - nm) + __expf(s - nm);
            rmax[r] = nm;
        }
    }

    // combine (max,sum) across the 16 lanes of each half (rows r+8g)
    for (int mask = 1; mask < 16; mask <<= 1) {
#pragma unroll
        for (int r = 0; r < 8; ++r) {
            const float om = __shfl_xor(rmax[r], mask, 32);
            const float os = __shfl_xor(rsum[r], mask, 32);
            const float nm = fmaxf(rmax[r], om);
            rsum[r] = rsum[r] * __expf(rmax[r] - nm) + os * __expf(om - nm);
            rmax[r] = nm;
        }
    }
    if (ml == 0) {
#pragma unroll
        for (int r = 0; r < 8; ++r) {
            rowmax[(size_t)bh * SEQ_T + i0 + r + 8 * g] = rmax[r];
            rowsum[(size_t)bh * SEQ_T + i0 + r + 8 * g] = rsum[r];
        }
    }
}

// ---------------------------------------------------------------------------
// Pass 2: normalize attn in place (p = exp(s-max)/sum) and compute P @ V via
// WMMA (A = p as bf16, B = Vt tiles). One wave per 16-row tile per (b,h).
// ---------------------------------------------------------------------------
__global__ __launch_bounds__(32)
void pv_pass2(float* __restrict__ attn, const unsigned short* __restrict__ vt,
              const float* __restrict__ rowmax, const float* __restrict__ rowsum,
              float* __restrict__ a2)
{
    const int bh   = blockIdx.x >> 7;
    const int i0   = (blockIdx.x & 127) * 16;
    const int lane = threadIdx.x;
    const int ml   = lane & 15;
    const int g    = lane >> 4;
    const int b = bh >> 4, h = bh & (N_HEADS - 1);

    float* arow = attn + ((size_t)bh * SEQ_T + i0 + ml) * SEQ_T;   // A-frag row m = ml
    const float rm = rowmax[(size_t)bh * SEQ_T + i0 + ml];
    const float rs = 1.f / rowsum[(size_t)bh * SEQ_T + i0 + ml];
    const unsigned short* vbase = vt + (size_t)bh * HEAD_DIM * SEQ_T;

    v8f c0 = {}, c1 = {}, c2 = {}, c3 = {};

    for (int kc = 0; kc < SEQ_T / 32; ++kc) {
        const int j0 = kc * 32;
        U8u au;
#pragma unroll
        for (int j = 0; j < 8; ++j) {
            const int col = j0 + kA_of(j, g);
            const float p0 = __expf(arow[col] - rm) * rs;
            const float p1 = __expf(arow[col + 1] - rm) * rs;
            arow[col] = p0; arow[col + 1] = p1;                 // normalized output
            au.u[j] = packbf(p0, p1);
        }
        const v16bf a = __builtin_bit_cast(v16bf, au.v);

        U8u bu;
#pragma unroll
        for (int dt = 0; dt < 4; ++dt) {
            const unsigned* vrow =
                (const unsigned*)(vbase + (size_t)(dt * 16 + ml) * SEQ_T);
#pragma unroll
            for (int j = 0; j < 8; ++j) bu.u[j] = vrow[(j0 + kB_of(j, g)) >> 1];
            const v16bf bb = __builtin_bit_cast(v16bf, bu.v);
            if (dt == 0) c0 = __builtin_amdgcn_wmma_f32_16x16x32_bf16(false, a, false, bb, (short)0, c0, false, false);
            if (dt == 1) c1 = __builtin_amdgcn_wmma_f32_16x16x32_bf16(false, a, false, bb, (short)0, c1, false, false);
            if (dt == 2) c2 = __builtin_amdgcn_wmma_f32_16x16x32_bf16(false, a, false, bb, (short)0, c2, false, false);
            if (dt == 3) c3 = __builtin_amdgcn_wmma_f32_16x16x32_bf16(false, a, false, bb, (short)0, c3, false, false);
        }
    }

    U8f u[4]; u[0].v = c0; u[1].v = c1; u[2].v = c2; u[3].v = c3;
#pragma unroll
    for (int dt = 0; dt < 4; ++dt)
#pragma unroll
        for (int r = 0; r < 8; ++r) {
            const int gm = b * SEQ_T + i0 + r + 8 * g;          // row in (B*T)
            const int gn = h * HEAD_DIM + dt * 16 + ml;
            a2[(size_t)gm * D_MODEL + gn] = u[dt].f[r];
        }
}

// ---------------------------------------------------------------------------
extern "C" void kernel_launch(void* const* d_in, const int* in_sizes, int n_in,
                              void* d_out, int out_size, void* d_ws, size_t ws_size,
                              hipStream_t stream)
{
    const float* x   = (const float*)d_in[0];
    const float* Wn  = (const float*)d_in[1];
    const float* Wv  = (const float*)d_in[2];
    const float* Wo  = (const float*)d_in[3];
    const float* aw0 = (const float*)d_in[4];

    const int BT = in_sizes[0] / D_MODEL;   // B*T = 4096
    const int Bb = BT / SEQ_T;              // B = 2

    float* out  = (float*)d_out;                              // (B*T, D) fp32
    float* attn = out + (size_t)BT * D_MODEL;                 // (B,H,T,T) fp32

    // workspace partition
    char* w = (char*)d_ws;
    float*          nodes_f32 = (float*)w;           w += (size_t)BT * D_MODEL * 4;
    unsigned short* nodes_bf  = (unsigned short*)w;  w += (size_t)BT * D_MODEL * 2;
    unsigned short* vtb       = (unsigned short*)w;  w += (size_t)BT * D_MODEL * 2;
    float*          axisb     = (float*)w;           w += (size_t)Bb * N_HEADS * SEQ_T * 4;
    float*          rmaxb     = (float*)w;           w += (size_t)Bb * N_HEADS * SEQ_T * 4;
    float*          rsumb     = (float*)w;           w += (size_t)Bb * N_HEADS * SEQ_T * 4;
    float*          a2        = (float*)w;           // (B*T, D) fp32 per-head attn out

    const dim3 blk(256);
    const dim3 grd(D_MODEL / 64, BT / 64);

    // nodes = x @ W_node^T   (fp32 + bf16 copies)
    proj_gemm_kernel<<<grd, blk, 0, stream>>>(x, Wn, nodes_f32, nodes_bf, nullptr,
                                              BT, D_MODEL, D_MODEL);
    // values = x @ W_value^T (bf16, per-head transposed Vt[b][h][d][t])
    proj_gemm_kernel<<<grd, blk, 0, stream>>>(x, Wv, nullptr, nullptr, vtb,
                                              BT, D_MODEL, D_MODEL);
    // axis = nodes . arity_w0 per head
    axis_kernel<<<(Bb * N_HEADS * SEQ_T) / 256, 256, 0, stream>>>(nodes_f32, aw0, axisb);

    const int nblk = Bb * N_HEADS * (SEQ_T / 16);
    scores_pass1<<<nblk, 32, 0, stream>>>(nodes_bf, axisb, attn, rmaxb, rsumb);
    pv_pass2<<<nblk, 32, 0, stream>>>(attn, vtb, rmaxb, rsumb, a2);

    // out = a2 @ W_out^T  (fp32)
    proj_gemm_kernel<<<grd, blk, 0, stream>>>(a2, Wo, out, nullptr, nullptr,
                                              BT, D_MODEL, D_MODEL);
}
